// DualPGD_3092376453437
// MI455X (gfx1250) — matrix operand order repaired and checked by reading, not compiled, not run on hardware
//
#include <hip/hip_runtime.h>
#include <math.h>

typedef __attribute__((ext_vector_type(16))) _Float16 v16h;
typedef __attribute__((ext_vector_type(8)))  float    v8f;

#define HD   256
#define IMG  (HD * HD)            // 65536
#define NB   64                   // batch
#define NIMG ((size_t)NB * IMG)   // 4194304 floats per image-buffer
#define TAU    0.25f
#define GAMMA  1.0f
#define NORM_STOP 1e-4f
#define ITERS 30

__device__ __forceinline__ v8f wmma16(v16h a, v16h b, v8f c) {
    // v_wmma_f32_16x16x32_f16: D = A(16x32 f16) * B(32x16 f16) + C(f32)
    return __builtin_amdgcn_wmma_f32_16x16x32_f16(
        /*neg_a=*/false, a, /*neg_b=*/false, b,
        /*c_mod=*/(short)0, c, /*reuse_a=*/false, /*reuse_b=*/false);
}

// ---------------------------------------------------------------------------
// Hadamard-sandwich GEMM, one wave per 16x16 output tile, K=256 in 8 steps.
// Split-f16: Hadamard entries (+-1/16) are exact in f16; only the data
// operand is split into hi+lo f16 and fed through two WMMA chains.
// MODE 1: C = H      * split(data)        (measure, left)
// MODE 2: C = split(data) * H^T           (measure, right)
// MODE 3: C = H^T    * split(data)        (adjoint, left)
// MODE 4: C = split(data) * H             (adjoint, right)
// EPI  0: out = acc
// EPI  1: out = acc - aux                  (res = measure(x) - m)
// EPI  2: out = 0.5*((aux - GAMMA*acc)+1)  (v after gradient step)
// ---------------------------------------------------------------------------
template <int MODE, int EPI>
__global__ __launch_bounds__(256)
void gemm_had(const float* __restrict__ Hm, const float* __restrict__ data,
              const float* __restrict__ aux, float* __restrict__ out)
{
    const int lane = threadIdx.x & 31;
    const int wave = threadIdx.x >> 5;
    const int l16  = lane & 15;
    const int lhi  = lane >> 4;

    const int gt   = blockIdx.x * 8 + wave;     // global tile id
    const int img  = gt >> 8;                   // 256 tiles per image
    const int tile = gt & 255;
    const int tm   = (tile >> 4) << 4;
    const int tn   = (tile & 15) << 4;

    const float* __restrict__ db = data + (size_t)img * IMG;

    v8f acc = {};
    for (int kb = 0; kb < HD; kb += 32) {
        if (MODE == 1 || MODE == 3) {
            // ---- A = H (exact f16), B = data (split hi/lo) ----
            v16h a0, bh, bl;
            if (MODE == 1) {
                // a[j] = H[tm+l16, kb + lhi*8 + j'], j'∈{0..7}, +16 for j>=8
                const float* Ar = Hm + (size_t)(tm + l16) * HD + kb + lhi * 8;
#pragma unroll
                for (int j = 0; j < 8; ++j) {
                    a0[j]     = (_Float16)Ar[j];
                    a0[8 + j] = (_Float16)Ar[16 + j];
                }
            } else { // MODE 3: A[m,k] = H[k,m]
#pragma unroll
                for (int j = 0; j < 8; ++j) {
                    a0[j]     = (_Float16)Hm[(size_t)(kb + lhi * 8 + j) * HD + tm + l16];
                    a0[8 + j] = (_Float16)Hm[(size_t)(kb + 16 + lhi * 8 + j) * HD + tm + l16];
                }
            }
            // b[j] = data[kb + lhi*8 + j', tn + l16], j'∈{0..7}, +16 for j>=8
#pragma unroll
            for (int j = 0; j < 16; ++j) {
                float v = db[(size_t)(kb + lhi * 8 + (j & 7) + ((j >> 3) << 4)) * HD + tn + l16];
                _Float16 h = (_Float16)v;
                bh[j] = h;
                bl[j] = (_Float16)(v - (float)h);
            }
            acc = wmma16(a0, bl, acc);
            acc = wmma16(a0, bh, acc);
        } else {
            // ---- A = data (split hi/lo), B = H (exact f16) ----
            v16h ah, al, b0;
            const float* Ar = db + (size_t)(tm + l16) * HD + kb + lhi * 8;
#pragma unroll
            for (int j = 0; j < 8; ++j) {
                float v0 = Ar[j];
                float v1 = Ar[16 + j];
                _Float16 h0 = (_Float16)v0, h1 = (_Float16)v1;
                ah[j] = h0;  ah[8 + j] = h1;
                al[j] = (_Float16)(v0 - (float)h0);
                al[8 + j] = (_Float16)(v1 - (float)h1);
            }
            if (MODE == 2) {
                // B[k,l] = H[l,k] -> k = lhi*8 + j', +16 for j>=8
                const float* Br = Hm + (size_t)(tn + l16) * HD + kb + lhi * 8;
#pragma unroll
                for (int j = 0; j < 16; ++j) b0[j] = (_Float16)Br[(j & 7) + ((j >> 3) << 4)];
            } else { // MODE 4: B = H
#pragma unroll
                for (int j = 0; j < 16; ++j)
                    b0[j] = (_Float16)Hm[(size_t)(kb + lhi * 8 + (j & 7) + ((j >> 3) << 4)) * HD + tn + l16];
            }
            acc = wmma16(al, b0, acc);
            acc = wmma16(ah, b0, acc);
        }
    }

    // C layout: VGPR i -> row tm + lhi*8 + i, col tn + l16
    float* __restrict__ ob = out + (size_t)img * IMG;
    const float* __restrict__ ax = (EPI != 0) ? aux + (size_t)img * IMG : nullptr;
#pragma unroll
    for (int i = 0; i < 8; ++i) {
        size_t idx = (size_t)(tm + lhi * 8 + i) * HD + tn + l16;
        float r = acc[i];
        if (EPI == 1)      r = r - ax[idx];
        else if (EPI == 2) { float xg = ax[idx] - GAMMA * r; r = 0.5f * (xg + 1.0f); }
        ob[idx] = r;
    }
}

// ---------------------------------------------------------------------------
// Dual update: u = clip(u + tau * grad2d(v)) with l = MU*GAMMA = 1.
// ---------------------------------------------------------------------------
__global__ void k_u(const float* __restrict__ V, float* __restrict__ U)
{
    size_t p = (size_t)blockIdx.x * blockDim.x + threadIdx.x;
    int img = (int)(p >> 16);
    int ij  = (int)(p & 65535);
    int i = ij >> 8, j = ij & 255;
    const float* v = V + (size_t)img * IMG;
    float vc = v[ij];
    float gx = (i < HD - 1) ? (v[ij + HD] - vc) : 0.0f;
    float gy = (j < HD - 1) ? (v[ij + 1]  - vc) : 0.0f;
    float* u0 = U + (size_t)img * 2 * IMG;
    float* u1 = u0 + IMG;
    float a = u0[ij] + TAU * gx;  a = a / fmaxf(1.0f, fabsf(a));
    float b = u1[ij] + TAU * gy;  b = b / fmaxf(1.0f, fabsf(b));
    u0[ij] = a;
    u1[ij] = b;
}

// ---------------------------------------------------------------------------
// Divergence + commit + norm accumulation (predicated on done flag).
// ---------------------------------------------------------------------------
__global__ void k_div(const float* __restrict__ V, const float* __restrict__ U,
                      float* __restrict__ X, float* __restrict__ norms,
                      const int* __restrict__ flag, int iter)
{
    __shared__ float s0[256];
    __shared__ float s1[256];
    if (*flag) return;   // uniform across the grid

    size_t p = (size_t)blockIdx.x * blockDim.x + threadIdx.x;
    int img = (int)(p >> 16);
    int ij  = (int)(p & 65535);
    int i = ij >> 8, j = ij & 255;
    const float* u0 = U + (size_t)img * 2 * IMG;
    const float* u1 = u0 + IMG;

    float axv = (i == 0) ? -u0[ij] : (u0[ij - HD] - u0[ij]);
    float ayv = (j == 0) ? -u1[ij] : (u1[ij - 1]  - u1[ij]);
    float xc  = 2.0f * (V[p] - (axv + ayv)) - 1.0f;
    float d   = xc - X[p];
    X[p] = xc;

    int t = threadIdx.x;
    s0[t] = d * d;
    s1[t] = xc * xc;
    __syncthreads();
#pragma unroll
    for (int s = 128; s > 0; s >>= 1) {
        if (t < s) { s0[t] += s0[t + s]; s1[t] += s1[t + s]; }
        __syncthreads();
    }
    if (t == 0) {
        atomicAdd(&norms[2 * iter],     s0[0]);
        atomicAdd(&norms[2 * iter + 1], s1[0]);
    }
}

__global__ void k_flag(const float* __restrict__ norms, int* __restrict__ flag, int iter)
{
    if (threadIdx.x == 0 && blockIdx.x == 0) {
        if (!*flag) {
            float a = norms[2 * iter], b = norms[2 * iter + 1];
            if (sqrtf(a) < NORM_STOP * sqrtf(b)) *flag = 1;
        }
    }
}

__global__ void k_init(float* __restrict__ U, float* __restrict__ norms,
                       int* __restrict__ flag, size_t uN)
{
    size_t p = (size_t)blockIdx.x * blockDim.x + threadIdx.x;
    if (p < uN) U[p] = 0.0f;
    if (p < 2 * ITERS) norms[p] = 0.0f;
    if (p == 0) *flag = 0;
}

// ---------------------------------------------------------------------------
extern "C" void kernel_launch(void* const* d_in, const int* in_sizes, int n_in,
                              void* d_out, int out_size, void* d_ws, size_t ws_size,
                              hipStream_t stream)
{
    const float* x_in = (const float*)d_in[0];   // (64,1,256,256) f32
    const float* Hm   = (const float*)d_in[1];   // (256,256) f32 Hadamard/sqrt(N)

    float* ws = (float*)d_ws;
    float* X  = ws;                // current x        (NIMG)
    float* M  = X  + NIMG;         // measurement m    (NIMG)
    float* W1 = M  + NIMG;         // ping             (NIMG)
    float* W2 = W1 + NIMG;         // pong / v         (NIMG)
    float* U  = W2 + NIMG;         // dual u, 2 planes (2*NIMG)
    float* norms = U + 2 * NIMG;   // 2 per iteration
    int*   flag  = (int*)(norms + 2 * ITERS);

    const dim3 gB((unsigned)(NIMG / IMG * 256 / 8));  // 64 img * 256 tiles / 8 waves = 2048
    const dim3 tB(256);
    const unsigned pxBlocks = (unsigned)(NIMG / 256);

    // Reset per-call device state (graph replays re-run this first).
    size_t uN = 2 * NIMG;
    k_init<<<(unsigned)((uN + 255) / 256), 256, 0, stream>>>(U, norms, flag, uN);

    // m = H x H^T   (linear-equivalent of 2*measure((x+1)/2) - measure(1))
    gemm_had<1, 0><<<gB, tB, 0, stream>>>(Hm, x_in, nullptr, W1);
    gemm_had<2, 0><<<gB, tB, 0, stream>>>(Hm, W1,  nullptr, M);
    // x0 = H^T m H
    gemm_had<3, 0><<<gB, tB, 0, stream>>>(Hm, M,   nullptr, W1);
    gemm_had<4, 0><<<gB, tB, 0, stream>>>(Hm, W1,  nullptr, X);

    for (int it = 0; it < ITERS; ++it) {
        gemm_had<1, 0><<<gB, tB, 0, stream>>>(Hm, X,  nullptr, W1); // T = H x
        gemm_had<2, 1><<<gB, tB, 0, stream>>>(Hm, W1, M,       W2); // R = T H^T - m
        gemm_had<3, 0><<<gB, tB, 0, stream>>>(Hm, W2, nullptr, W1); // S = H^T R
        gemm_had<4, 2><<<gB, tB, 0, stream>>>(Hm, W1, X,       W2); // v = 0.5*((x - S H)+1)
        k_u  <<<pxBlocks, 256, 0, stream>>>(W2, U);
        k_div<<<pxBlocks, 256, 0, stream>>>(W2, U, X, norms, flag, it);
        k_flag<<<1, 32, 0, stream>>>(norms, flag, it);
    }

    hipMemcpyAsync(d_out, X, (size_t)out_size * sizeof(float),
                   hipMemcpyDeviceToDevice, stream);
}
